// ConditionalRandomField_69432441307207
// MI455X (gfx1250) — matrix operand order repaired and checked by reading, not compiled, not run on hardware
//
#include <hip/hip_runtime.h>
#include <hip/hip_bf16.h>
#include <stdint.h>

#define NT 256   // num tags
#define BB 64    // batch
#define SS 1024  // seq len
#define NW 16    // waves per forward block (16 waves * 16 cols = 256 cols)

typedef __attribute__((ext_vector_type(16))) __bf16 v16bf;
typedef __attribute__((ext_vector_type(8)))  float  v8f;

union U16bf { v16bf v; unsigned int u[8]; };

__device__ __forceinline__ unsigned int f32_to_bf16_rne(float x) {
    unsigned int b = __float_as_uint(x);
    unsigned int r = b + 0x7FFFu + ((b >> 16) & 1u);
    return r >> 16;
}
__device__ __forceinline__ float bf16_to_f32(unsigned int h) {
    return __uint_as_float(h << 16);
}

// gfx1250 async copy global -> LDS (per-lane b32), tracked by ASYNCcnt.
__device__ __forceinline__ void async_ld_b32(void* ldsdst, const void* gbase, unsigned goff) {
    unsigned lds = (unsigned)(uintptr_t)ldsdst;          // low 32 bits of flat addr = LDS offset
    unsigned long long sb = (unsigned long long)(uintptr_t)gbase;
    asm volatile("global_load_async_to_lds_b32 %0, %1, %2"
                 :: "v"(lds), "v"(goff), "s"(sb) : "memory");
}
__device__ __forceinline__ void wait_async0() {
    asm volatile("s_wait_asynccnt 0x0" ::: "memory");
}

// ---------------------------------------------------------------------------
// Forward (log-partition) kernel: one block = one batch chain.
// alpha'[j] = m + log( exp(alpha-m) . exp(T) )[j] + emit[t,j]
// exp(T) kept in VGPRs as bf16 hi/lo split; 24 v_wmma_f32_16x16x32_bf16 / step
// in 3 independent accumulator chains (hi*Bhi, lo*Bhi, hi*Blo).
// ---------------------------------------------------------------------------
__global__ void __launch_bounds__(NW * 32)
crf_forward(const float* __restrict__ inputs, const unsigned char* __restrict__ mask,
            const float* __restrict__ trans, const float* __restrict__ start,
            const float* __restrict__ stop, float* __restrict__ logZ) {
    const int tid   = threadIdx.x;
    const int lane  = tid & 31;
    const int w     = tid >> 5;       // wave id 0..15
    const int row   = lane & 15;      // A row / B,D column index this lane represents
    const int khalf = lane >> 4;      // which K half-group this lane holds
    const int col   = (w << 4) + row; // this lane's tag column (dup on lanes 16-31)
    const int b     = blockIdx.x;

    const float* emitb = inputs + (size_t)b * SS * NT;
    const unsigned char* maskb = mask + (size_t)b * SS;

    __shared__ unsigned short vhiL[NT];   // 512 B: v hi-part, A-layout source
    __shared__ unsigned short vloL[NT];   // 512 B: v lo-part
    __shared__ unsigned int   zeroL[128]; // 512 B of zeros: A rows 1..15 read here
    __shared__ float ebuf[2][NT];
    __shared__ float wmx[2][NW];
    __shared__ float psum[NW];

    // zero region init (once); published by the first in-loop barrier
    if (tid < 128) zeroL[tid] = 0u;

    // ---- load B = exp(T) tile into registers, bf16 hi/lo split, WMMA B layout.
    // 16-bit B 32x16, this code's k order (A matches): lanes 0-15 hold K=0..15 (2/dword), lanes 16-31 K=16..31.
    U16bf Bh[8], Bl[8];
#pragma unroll
    for (int kc = 0; kc < 8; ++kc) {
#pragma unroll
        for (int j = 0; j < 8; ++j) {
            int k0 = kc * 32 + khalf * 16 + 2 * j;
            float e0 = __expf(trans[k0 * NT + col]);
            float e1 = __expf(trans[(k0 + 1) * NT + col]);
            unsigned h0 = f32_to_bf16_rne(e0), h1 = f32_to_bf16_rne(e1);
            float l0 = e0 - bf16_to_f32(h0), l1 = e1 - bf16_to_f32(h1);
            Bh[kc].u[j] = h0 | (h1 << 16);
            Bl[kc].u[j] = f32_to_bf16_rne(l0) | (f32_to_bf16_rne(l1) << 16);
        }
    }

    // A-source base pointers: row 0 lanes read the real v arrays, others read zeros.
    // dword index per chunk kc: kc*16 + khalf*8 + {0..7}
    const bool r0 = (row == 0);
    const unsigned int* aph = (r0 ? (const unsigned int*)vhiL : (const unsigned int*)zeroL) + khalf * 8;
    const unsigned int* apl = (r0 ? (const unsigned int*)vloL : (const unsigned int*)zeroL) + khalf * 8;

    // ---- alpha0 = start + emissions[t=0]
    float alpha = start[col] + emitb[col];

    // wave-local max of this wave's 16 columns (halves are duplicates)
    {
        float mv = (khalf == 0) ? alpha : -__builtin_inff();
#pragma unroll
        for (int d = 1; d < 32; d <<= 1) mv = fmaxf(mv, __shfl_xor(mv, d, 32));
        if (lane == 0) wmx[0][w] = mv;
    }

    // prime async emission row t=1 into ebuf[1]
    if (tid < NT) async_ld_b32(&ebuf[1][tid], emitb, (unsigned)((1u * NT + tid) * 4u));

#pragma unroll 1
    for (int t = 1; t < SS; ++t) {
        wait_async0();
        __syncthreads();                 // B1: wmx[(t-1)&1] + ebuf[t&1] visible

        float m = wmx[(t - 1) & 1][0];
#pragma unroll
        for (int i = 1; i < NW; ++i) m = fmaxf(m, wmx[(t - 1) & 1][i]);

        float v = __expf(alpha - m);     // lanes 16-31: exp(-inf)=0, harmless dup
        unsigned hv = f32_to_bf16_rne(v);
        unsigned lv = f32_to_bf16_rne(v - bf16_to_f32(hv));
        if (khalf == 0) { vhiL[col] = (unsigned short)hv; vloL[col] = (unsigned short)lv; }
        __syncthreads();                 // B2: v published

        // ---- A fragments come straight from LDS (zero rows via zeroL base),
        //      3 independent WMMA accumulator chains for ILP.
        v8f c0 = {}, c1 = {}, c2 = {};
#pragma unroll
        for (int kc = 0; kc < 8; ++kc) {
            U16bf Ah, Al;
#pragma unroll
            for (int j = 0; j < 8; ++j) {
                int idx = kc * 16 + j;   // {0..7}
                Ah.u[j] = aph[idx];
                Al.u[j] = apl[idx];
            }
            c0 = __builtin_amdgcn_wmma_f32_16x16x32_bf16(false, Ah.v, false, Bh[kc].v, (short)0, c0, false, false);
            c1 = __builtin_amdgcn_wmma_f32_16x16x32_bf16(false, Al.v, false, Bh[kc].v, (short)0, c1, false, false);
            c2 = __builtin_amdgcn_wmma_f32_16x16x32_bf16(false, Ah.v, false, Bl[kc].v, (short)0, c2, false, false);
        }

        float wsum = (c0[0] + c1[0]) + c2[0];   // D row 0, N=lane (lanes 0-15)
        float cand = m + __logf(wsum) + ebuf[t & 1][col];
        alpha = maskb[t] ? cand : alpha;

        // tail: publish this step's wave max into the other parity slot
        float mv = (khalf == 0) ? alpha : -__builtin_inff();
#pragma unroll
        for (int d = 1; d < 32; d <<= 1) mv = fmaxf(mv, __shfl_xor(mv, d, 32));
        if (lane == 0) wmx[t & 1][w] = mv;

        if (t + 1 < SS && tid < NT)
            async_ld_b32(&ebuf[(t + 1) & 1][tid], emitb, (unsigned)(((unsigned)(t + 1) * NT + tid) * 4u));
    }

    // ---- logZ[b] = logsumexp(alpha + stop)
    __syncthreads();
    float aj = alpha + stop[col];
    float mv = (khalf == 0) ? aj : -__builtin_inff();
#pragma unroll
    for (int d = 1; d < 32; d <<= 1) mv = fmaxf(mv, __shfl_xor(mv, d, 32));
    if (lane == 0) wmx[0][w] = mv;
    __syncthreads();
    float m = wmx[0][0];
#pragma unroll
    for (int i = 1; i < NW; ++i) m = fmaxf(m, wmx[0][i]);
    float e = (khalf == 0) ? __expf(aj - m) : 0.f;
#pragma unroll
    for (int d = 1; d < 32; d <<= 1) e += __shfl_xor(e, d, 32);
    if (lane == 0) psum[w] = e;
    __syncthreads();
    if (tid == 0) {
        float s = 0.f;
        for (int i = 0; i < NW; ++i) s += psum[i];
        logZ[b] = m + __logf(s);
    }
}

// ---------------------------------------------------------------------------
// Numerator (gold-path score) + final output: out[b] = score[b] - logZ[b]
// ---------------------------------------------------------------------------
__global__ void __launch_bounds__(256)
crf_score(const float* __restrict__ inputs, const int* __restrict__ tags,
          const unsigned char* __restrict__ mask, const float* __restrict__ trans,
          const float* __restrict__ start, const float* __restrict__ stop,
          const float* __restrict__ logZ, float* __restrict__ out) {
    const int b = blockIdx.x, tid = threadIdx.x;
    const int* tg = tags + (size_t)b * SS;
    const unsigned char* mk = mask + (size_t)b * SS;
    const float* em = inputs + (size_t)b * SS * NT;

    float acc = 0.f;
    for (int t = tid; t < SS; t += 256) {
        int tag = tg[t];
        float mft = mk[t] ? 1.f : 0.f;
        if (t >= 1)      acc += trans[tg[t - 1] * NT + tag] * mft;
        if (t <= SS - 2) acc += em[(size_t)t * NT + tag] * mft;
    }
    if (tid == 0)   acc += start[tg[0]];
    if (tid == 255) acc += (stop[tg[SS - 1]] + em[(size_t)(SS - 1) * NT + tg[SS - 1]]) *
                           (mk[SS - 1] ? 1.f : 0.f);

#pragma unroll
    for (int d = 1; d < 32; d <<= 1) acc += __shfl_xor(acc, d, 32);
    __shared__ float part[8];
    if ((tid & 31) == 0) part[tid >> 5] = acc;
    __syncthreads();
    if (tid == 0) {
        float s = 0.f;
        for (int i = 0; i < 8; ++i) s += part[i];
        out[b] = s - logZ[b];
    }
}

extern "C" void kernel_launch(void* const* d_in, const int* in_sizes, int n_in,
                              void* d_out, int out_size, void* d_ws, size_t ws_size,
                              hipStream_t stream) {
    const float*         inputs = (const float*)d_in[0];
    const int*           tags   = (const int*)d_in[1];
    const unsigned char* mask   = (const unsigned char*)d_in[2];
    const float*         trans  = (const float*)d_in[3];
    const float*         start  = (const float*)d_in[4];
    const float*         stop   = (const float*)d_in[5];
    float* out  = (float*)d_out;
    float* logZ = (float*)d_ws;   // 64 floats of scratch

    crf_forward<<<BB, NW * 32, 0, stream>>>(inputs, mask, trans, start, stop, logZ);
    crf_score<<<BB, 256, 0, stream>>>(inputs, tags, mask, trans, start, stop, logZ, out);
}
